// GCN_52329881534967
// MI455X (gfx1250) — matrix operand order repaired and checked by reading, not compiled, not run on hardware
//
#include <hip/hip_runtime.h>
#include <hip/hip_bf16.h>
#include <stdint.h>

// ---------------------------------------------------------------------------
// GCN forward for MI455X (gfx1250).
// Strategy: bf16 feature matrices (L2-resident, halves edge traffic),
// v_wmma_f32_16x16x32_bf16 for dense transforms (fp32 accumulate),
// global_atomic_pk_add_bf16 for the edge scatter-add (2 feats / atomic),
// fp32 everywhere sizes are tiny (degrees, pooling, FFN).
// ---------------------------------------------------------------------------

#define DIM 128          // node / hidden dim
#define GRAPHS 128
#define FFN 512
#define CLASSES 10

typedef __attribute__((ext_vector_type(16))) __bf16 v16bf;
typedef __attribute__((ext_vector_type(8)))  float  v8f;

union ABu { uint4 u[2]; v16bf v; };

__device__ __forceinline__ float bf2f(unsigned int lo16) {
  return __uint_as_float(lo16 << 16);
}
__device__ __forceinline__ unsigned int f2bf(float f) {  // round-to-nearest-even
  unsigned int u = __float_as_uint(f);
  return (u + 0x7FFFu + ((u >> 16) & 1u)) >> 16;
}
__device__ __forceinline__ void pk_atomic_bf16(unsigned short* p, unsigned int d) {
  unsigned long long a = (unsigned long long)p;
  asm volatile("global_atomic_pk_add_bf16 %0, %1, off scope:SCOPE_DEV"
               :: "v"(a), "v"(d) : "memory");
}

// ---------------- degree / norm ----------------
__global__ void k_deg_init(float* deg, int N) {
  int i = blockIdx.x * 256 + threadIdx.x;
  if (i < N) deg[i] = 1.0f;                       // self-loop contributes 1
}
__global__ void k_deg_accum(const int* __restrict__ col, float* deg, int E) {
  int e = blockIdx.x * 256 + threadIdx.x;
  if (e < E) atomicAdd(&deg[col[e]], 1.0f);
}
__global__ void k_dinv(float* deg, int N) {
  int i = blockIdx.x * 256 + threadIdx.x;
  if (i < N) deg[i] = rsqrtf(deg[i]);             // deg >= 1 always
}

// ---------------- fp32 -> bf16 feature convert ----------------
__global__ void k_cvt_x(const float* __restrict__ x, unsigned short* __restrict__ hb, int total4) {
  int t = blockIdx.x * 256 + threadIdx.x;
  if (t >= total4) return;
  float4 v = *(const float4*)(x + (size_t)t * 4);
  unsigned int lo = f2bf(v.x) | (f2bf(v.y) << 16);
  unsigned int hi = f2bf(v.z) | (f2bf(v.w) << 16);
  *(unsigned long long*)(hb + (size_t)t * 4) =
      (unsigned long long)lo | ((unsigned long long)hi << 32);
}

// ---------------- pack W (128x128 f32 row-major) into WMMA B-operand order ---
// Layout per 32-K tile: VGPR v, lanes 0-15: K=2v,2v+1 ; lanes 16-31: K=16+2v,+1
// Bp dword index = ((ntile*4 + ktile)*32 + lane)*8 + v
__global__ void k_packW(const float* __restrict__ W, unsigned int* __restrict__ Bp) {
  int t = blockIdx.x * 256 + threadIdx.x;            // 0 .. 8191
  if (t >= 8 * 4 * 32 * 8) return;
  int v     = t & 7;
  int lane  = (t >> 3) & 31;
  int ktile = (t >> 8) & 3;
  int ntile = t >> 10;
  int k = ktile * 32 + (lane >> 4) * 16 + 2 * v;
  int n = ntile * 16 + (lane & 15);
  unsigned int lo = f2bf(W[(size_t)k * DIM + n]);
  unsigned int hi = f2bf(W[(size_t)(k + 1) * DIM + n]);
  Bp[t] = lo | (hi << 16);
}

// ---------------- t = hb @ W via WMMA bf16 (fp32 accum) ----------------
// One wave per 16x16 tile of C; block of 8 waves covers one 16-row strip.
__global__ void k_gemm(const unsigned short* __restrict__ hb,
                       const unsigned int*  __restrict__ Bp,
                       unsigned short*      __restrict__ t) {
  const int lane = threadIdx.x & 31;
  const int wave = threadIdx.x >> 5;         // = ntile (0..7)
  const int m0   = blockIdx.x * 16;
  const int half = lane >> 4;
  const int mr   = lane & 15;
  const unsigned short* arow = hb + (size_t)(m0 + mr) * DIM;
  v8f c = {};
#pragma unroll
  for (int kt = 0; kt < 4; ++kt) {
    ABu a, b;
    const int ka = kt * 32 + half * 16;
    a.u[0] = *(const uint4*)(arow + ka);          // K = ka .. ka+7
    a.u[1] = *(const uint4*)(arow + ka + 8);      // K = ka+8 .. ka+15
    const unsigned int* bp = Bp + (((size_t)(wave * 4 + kt) * 32 + lane) * 8);
    b.u[0] = *(const uint4*)(bp);
    b.u[1] = *(const uint4*)(bp + 4);
    c = __builtin_amdgcn_wmma_f32_16x16x32_bf16(false, a.v, false, b.v,
                                                (short)0, c, false, false);
  }
  const int n = wave * 16 + mr;
  unsigned short* trow = t + (size_t)(m0 + half * 8) * DIM + n;
#pragma unroll
  for (int v = 0; v < 8; ++v)
    trow[(size_t)v * DIM] = (unsigned short)f2bf(c[v]);
}

// ---------------- edge scatter: agg[col] += t[row] * dinv[row]*dinv[col] -----
// One wave per edge; lane handles 4 features (b64 gather, 2 pk-bf16 atomics).
__global__ void k_agg(const unsigned short* __restrict__ t,
                      const int* __restrict__ rowI, const int* __restrict__ colI,
                      const float* __restrict__ dinv,
                      unsigned short* __restrict__ agg, int E) {
  const int e = blockIdx.x * 8 + (threadIdx.x >> 5);
  if (e >= E) return;
  const int lane = threadIdx.x & 31;
  const int r = rowI[e];
  const int c = colI[e];
  const float nrm = dinv[r] * dinv[c];
  unsigned long long v =
      *(const unsigned long long*)(t + (size_t)r * DIM + lane * 4);
  unsigned int p0 = (unsigned int)v, p1 = (unsigned int)(v >> 32);
  float f0 = bf2f(p0 & 0xffffu) * nrm, f1 = bf2f(p0 >> 16) * nrm;
  float f2 = bf2f(p1 & 0xffffu) * nrm, f3 = bf2f(p1 >> 16) * nrm;
  unsigned int q0 = f2bf(f0) | (f2bf(f1) << 16);
  unsigned int q1 = f2bf(f2) | (f2bf(f3) << 16);
  unsigned short* dst = agg + (size_t)c * DIM + lane * 4;
  pk_atomic_bf16(dst, q0);
  pk_atomic_bf16(dst + 2, q1);
}

// ---------------- epilogue: hb = relu(agg + t*dinv^2 + bias) ----------------
__global__ void k_epi(const unsigned short* __restrict__ agg,
                      const unsigned short* __restrict__ t,
                      const float* __restrict__ dinv,
                      const float* __restrict__ bias,
                      unsigned short* __restrict__ hb, int total4) {
  int tid = blockIdx.x * 256 + threadIdx.x;
  if (tid >= total4) return;
  int node = tid >> 5;
  int j = (tid & 31) * 4;
  float sd = dinv[node]; sd *= sd;
  unsigned long long av = *(const unsigned long long*)(agg + (size_t)node * DIM + j);
  unsigned long long tv = *(const unsigned long long*)(t + (size_t)node * DIM + j);
  float4 bb = *(const float4*)(bias + j);
  float o0 = fmaxf(bf2f((unsigned int)av & 0xffffu) + bf2f((unsigned int)tv & 0xffffu) * sd + bb.x, 0.0f);
  float o1 = fmaxf(bf2f(((unsigned int)av >> 16) & 0xffffu) + bf2f(((unsigned int)tv >> 16) & 0xffffu) * sd + bb.y, 0.0f);
  float o2 = fmaxf(bf2f((unsigned int)(av >> 32) & 0xffffu) + bf2f((unsigned int)(tv >> 32) & 0xffffu) * sd + bb.z, 0.0f);
  float o3 = fmaxf(bf2f((unsigned int)(av >> 48)) + bf2f((unsigned int)(tv >> 48)) * sd + bb.w, 0.0f);
  unsigned int lo = f2bf(o0) | (f2bf(o1) << 16);
  unsigned int hi = f2bf(o2) | (f2bf(o3) << 16);
  *(unsigned long long*)(hb + (size_t)node * DIM + j) =
      (unsigned long long)lo | ((unsigned long long)hi << 32);
}

// ---------------- global mean pool ----------------
__global__ void k_pool(const unsigned short* __restrict__ hb,
                       const int* __restrict__ batch,
                       float* pooled, float* cnt, int N) {
  int node = blockIdx.x * 8 + (threadIdx.x >> 5);
  if (node >= N) return;
  int lane = threadIdx.x & 31;
  int g = batch[node];
  unsigned long long v =
      *(const unsigned long long*)(hb + (size_t)node * DIM + lane * 4);
  float* pg = pooled + (size_t)g * DIM + lane * 4;
  atomicAdd(pg + 0, bf2f((unsigned int)v & 0xffffu));
  atomicAdd(pg + 1, bf2f(((unsigned int)v >> 16) & 0xffffu));
  atomicAdd(pg + 2, bf2f((unsigned int)(v >> 32) & 0xffffu));
  atomicAdd(pg + 3, bf2f((unsigned int)(v >> 48)));
  if (lane == 0) atomicAdd(&cnt[g], 1.0f);
}
__global__ void k_pooldiv(float* pooled, const float* cnt) {
  int t = blockIdx.x * 256 + threadIdx.x;
  if (t < GRAPHS * DIM) pooled[t] /= fmaxf(cnt[t >> 7], 1.0f);
}

// ---------------- small f32 FFN GEMM: out = (relu)(A@W + b) ----------------
__global__ void k_ffn(const float* __restrict__ A, const float* __restrict__ W,
                      const float* __restrict__ bias, float* __restrict__ out,
                      int M, int K, int Nout, int relu) {
  int t = blockIdx.x * 256 + threadIdx.x;
  if (t >= M * Nout) return;
  int m = t / Nout, n = t - m * Nout;
  const float* a = A + (size_t)m * K;
  float acc = bias[n];
  for (int k = 0; k < K; ++k) acc = fmaf(a[k], W[(size_t)k * Nout + n], acc);
  if (relu) acc = fmaxf(acc, 0.0f);
  out[t] = acc;
}

extern "C" void kernel_launch(void* const* d_in, const int* in_sizes, int n_in,
                              void* d_out, int out_size, void* d_ws, size_t ws_size,
                              hipStream_t stream) {
  (void)n_in; (void)out_size; (void)ws_size;
  const float* x     = (const float*)d_in[0];
  const int*   ei    = (const int*)d_in[1];     // int32 (JAX demotes int64 w/o x64)
  const int*   batch = (const int*)d_in[2];
  const float* Wl[3] = {(const float*)d_in[3], (const float*)d_in[5], (const float*)d_in[7]};
  const float* bl[3] = {(const float*)d_in[4], (const float*)d_in[6], (const float*)d_in[8]};
  const float* Wf1 = (const float*)d_in[9],  *bf1 = (const float*)d_in[10];
  const float* Wf2 = (const float*)d_in[11], *bf2 = (const float*)d_in[12];
  const float* Wf3 = (const float*)d_in[13], *bf3 = (const float*)d_in[14];

  const int N = in_sizes[0] / DIM;     // 100000 (divisible by 16)
  const int E = in_sizes[1] / 2;       // 1600000
  const int* rowI = ei;
  const int* colI = ei + E;

  // carve workspace (256B aligned)
  char* w = (char*)d_ws;
  auto carve = [&](size_t bytes) { void* p = (void*)w; w += (bytes + 255) & ~(size_t)255; return p; };
  float*          dinv   = (float*)carve((size_t)N * 4);
  unsigned short* hb     = (unsigned short*)carve((size_t)N * DIM * 2);
  unsigned short* tbuf   = (unsigned short*)carve((size_t)N * DIM * 2);
  unsigned short* agg    = (unsigned short*)carve((size_t)N * DIM * 2);
  unsigned int*   Bp     = (unsigned int*)carve((size_t)3 * 8192 * 4);
  float*          pooled = (float*)carve((size_t)GRAPHS * DIM * 4);
  float*          cnt    = (float*)carve((size_t)GRAPHS * 4);
  float*          z1     = (float*)carve((size_t)GRAPHS * FFN * 4);
  float*          z2     = (float*)carve((size_t)GRAPHS * FFN * 4);

  // degrees -> dinv (in place)
  k_deg_init<<<(N + 255) / 256, 256, 0, stream>>>(dinv, N);
  k_deg_accum<<<(E + 255) / 256, 256, 0, stream>>>(colI, dinv, E);
  k_dinv<<<(N + 255) / 256, 256, 0, stream>>>(dinv, N);

  // x -> bf16, pack weights once
  k_cvt_x<<<(N * 32 + 255) / 256, 256, 0, stream>>>(x, hb, N * 32);
  for (int l = 0; l < 3; ++l)
    k_packW<<<32, 256, 0, stream>>>(Wl[l], Bp + (size_t)l * 8192);

  // 3 GCN layers
  for (int l = 0; l < 3; ++l) {
    k_gemm<<<N / 16, 256, 0, stream>>>(hb, Bp + (size_t)l * 8192, tbuf);
    hipMemsetAsync(agg, 0, (size_t)N * DIM * 2, stream);
    k_agg<<<(E + 7) / 8, 256, 0, stream>>>(tbuf, rowI, colI, dinv, agg, E);
    k_epi<<<(N * 32 + 255) / 256, 256, 0, stream>>>(agg, tbuf, dinv, bl[l], hb, N * 32);
  }

  // pool + FFN
  hipMemsetAsync(pooled, 0, (size_t)GRAPHS * DIM * 4, stream);
  hipMemsetAsync(cnt, 0, (size_t)GRAPHS * 4, stream);
  k_pool<<<(N + 7) / 8, 256, 0, stream>>>(hb, batch, pooled, cnt, N);
  k_pooldiv<<<(GRAPHS * DIM + 255) / 256, 256, 0, stream>>>(pooled, cnt);
  k_ffn<<<(GRAPHS * FFN + 255) / 256, 256, 0, stream>>>(pooled, Wf1, bf1, z1, GRAPHS, DIM, FFN, 1);
  k_ffn<<<(GRAPHS * FFN + 255) / 256, 256, 0, stream>>>(z1, Wf2, bf2, z2, GRAPHS, FFN, FFN, 1);
  k_ffn<<<(GRAPHS * CLASSES + 255) / 256, 256, 0, stream>>>(z2, Wf3, bf3, (float*)d_out, GRAPHS, FFN, CLASSES, 0);
}
